// CascadedBranch_35218731828020
// MI455X (gfx1250) — matrix operand order repaired and checked by reading, not compiled, not run on hardware
//
#include <hip/hip_runtime.h>
#include <hip/hip_bf16.h>
#include <cstdint>

typedef __attribute__((ext_vector_type(16))) __bf16 v16bf;
typedef __attribute__((ext_vector_type(8)))  float  v8f;

#define D_AUDIO 768
#define D_TEXT  512
#define NROWS   2048        // B*K = 256*8
#define VOCAB   49408
#define NTILES  386         // VOCAB / 128
#define NSPLIT  4
#define MTILE   32          // query rows per block
#define VQ_SCALE 14.4269504088896f   // (1/0.1) * log2(e)

__device__ __forceinline__ unsigned short f2bf(float f) {
  unsigned int u = __float_as_uint(f);
  u += 0x7FFFu + ((u >> 16) & 1u);   // round-to-nearest-even
  return (unsigned short)(u >> 16);
}

union FragBF {
  v16bf v;
  uint4 q[2];
};

#define V8F_ZERO (v8f){0.f, 0.f, 0.f, 0.f, 0.f, 0.f, 0.f, 0.f}

// ---------------- kernel 1: proj = audio @ W + b  (1.6 GFLOP, VALU f32) ----
__global__ void proj_kernel(const float* __restrict__ audio,
                            const float* __restrict__ W,
                            const float* __restrict__ bias,
                            float* __restrict__ proj) {
  __shared__ float a[D_AUDIO];
  const int row = blockIdx.x;
  const int tid = threadIdx.x;
  const float* arow = audio + (size_t)row * D_AUDIO;
  for (int k = tid; k < D_AUDIO; k += 256) a[k] = arow[k];
  __syncthreads();
  for (int d = tid; d < D_TEXT; d += 256) {
    float acc = bias[d];
#pragma unroll 4
    for (int k = 0; k < D_AUDIO; ++k)
      acc = fmaf(a[k], W[(size_t)k * D_TEXT + d], acc);
    proj[(size_t)row * D_TEXT + d] = acc;
  }
}

// ---------------- kernel 2: per-feature mean / inv-std over 2048 rows ------
__global__ void bnstats_kernel(const float* __restrict__ proj,
                               float* __restrict__ mean,
                               float* __restrict__ invstd) {
  const int d = blockIdx.x;
  const int tid = threadIdx.x;
  float s = 0.f, s2 = 0.f;
  for (int r = tid; r < NROWS; r += 256) {
    float v = proj[(size_t)r * D_TEXT + d];
    s += v; s2 += v * v;
  }
  for (int off = 16; off > 0; off >>= 1) {
    s  += __shfl_xor(s,  off, 32);
    s2 += __shfl_xor(s2, off, 32);
  }
  __shared__ float ls[8], ls2[8];
  const int lane = tid & 31, w = tid >> 5;
  if (lane == 0) { ls[w] = s; ls2[w] = s2; }
  __syncthreads();
  if (tid == 0) {
    float S = 0.f, S2 = 0.f;
    for (int i = 0; i < 8; ++i) { S += ls[i]; S2 += ls2[i]; }
    float m   = S * (1.f / NROWS);
    float var = S2 * (1.f / NROWS) - m * m;
    mean[d]   = m;
    invstd[d] = rsqrtf(var + 1e-5f);
  }
}

// ---------------- kernel 3: BN affine + row-normalize -> Q (bf16) ----------
__global__ void qprep_kernel(const float* __restrict__ proj,
                             const float* __restrict__ mean,
                             const float* __restrict__ invstd,
                             const float* __restrict__ gamma,
                             const float* __restrict__ beta,
                             unsigned short* __restrict__ Qbf) {
  const int row = blockIdx.x, tid = threadIdx.x;
  const float* p = proj + (size_t)row * D_TEXT;
  const int d0 = tid, d1 = tid + 256;
  float x0 = (p[d0] - mean[d0]) * invstd[d0] * gamma[d0] + beta[d0];
  float x1 = (p[d1] - mean[d1]) * invstd[d1] * gamma[d1] + beta[d1];
  float s = x0 * x0 + x1 * x1;
  for (int off = 16; off > 0; off >>= 1) s += __shfl_xor(s, off, 32);
  __shared__ float ls[8];
  __shared__ float rnorm;
  const int lane = tid & 31, w = tid >> 5;
  if (lane == 0) ls[w] = s;
  __syncthreads();
  if (tid == 0) {
    float S = 0.f;
    for (int i = 0; i < 8; ++i) S += ls[i];
    rnorm = rsqrtf(S);
  }
  __syncthreads();
  const float rn = rnorm;
  unsigned short* q = Qbf + (size_t)row * D_TEXT;
  q[d0] = f2bf(x0 * rn);
  q[d1] = f2bf(x1 * rn);
}

// ---------------- kernel 4: row-normalize vocab -> K (bf16, row-major) -----
__global__ void kprep_kernel(const float* __restrict__ emb,
                             unsigned short* __restrict__ Kbf) {
  const int row = blockIdx.x, tid = threadIdx.x;
  const float* e = emb + (size_t)row * D_TEXT;
  const int d0 = tid, d1 = tid + 256;
  float x0 = e[d0], x1 = e[d1];
  float s = x0 * x0 + x1 * x1;
  for (int off = 16; off > 0; off >>= 1) s += __shfl_xor(s, off, 32);
  __shared__ float ls[8];
  __shared__ float rnorm;
  const int lane = tid & 31, w = tid >> 5;
  if (lane == 0) ls[w] = s;
  __syncthreads();
  if (tid == 0) {
    float S = 0.f;
    for (int i = 0; i < 8; ++i) S += ls[i];
    rnorm = rsqrtf(S);
  }
  __syncthreads();
  const float rn = rnorm;
  unsigned short* k = Kbf + (size_t)row * D_TEXT;
  k[d0] = f2bf(x0 * rn);
  k[d1] = f2bf(x1 * rn);
}

// ---------------- kernel 5: transpose raw emb -> Vt bf16 [512][49408] ------
__global__ void vt_kernel(const float* __restrict__ emb,
                          unsigned short* __restrict__ Vt) {
  __shared__ unsigned short tile[64][65];
  const int r0 = blockIdx.x * 64;   // vocab
  const int c0 = blockIdx.y * 64;   // feature
  const int tid = threadIdx.x;
#pragma unroll
  for (int l = 0; l < 16; ++l) {
    int e = tid + l * 256;
    int i = e >> 6, j = e & 63;
    tile[j][i] = f2bf(emb[(size_t)(r0 + i) * D_TEXT + c0 + j]);
  }
  __syncthreads();
#pragma unroll
  for (int l = 0; l < 16; ++l) {
    int e = tid + l * 256;
    int dj = e >> 6, ri = e & 63;
    Vt[(size_t)(c0 + dj) * VOCAB + r0 + ri] = tile[dj][ri];
  }
}

// ---------------- kernel 6: streaming softmax-attention over the vocab -----
// Block: 256 threads (8 waves), 32 query rows, interleaved vocab chunk.
// Per 128-vocab tile:
//   phase 1: each wave computes a 32x16 S region (2 WMMA per K-frag, K-frag
//            double-buffered from global), exp -> P tile in LDS
//   phase 2: each wave owns a 64-wide D-chunk; 8 P-frags hoisted to regs,
//            each V-frag feeds 2 WMMA (row halves).
__global__ __launch_bounds__(256)
void attn_kernel(const unsigned short* __restrict__ Qbf,
                 const unsigned short* __restrict__ Kbf,
                 const unsigned short* __restrict__ Vt,
                 float* __restrict__ numer,
                 float* __restrict__ denom) {
  __shared__ unsigned short qt[MTILE * D_TEXT];   // 32 KB Q tile
  __shared__ unsigned short ptile[MTILE * 128];   // 8 KB  P tile (bf16)
  __shared__ float dsum[MTILE];

  const int tid  = threadIdx.x;
  const int lane = tid & 31;
  const int w    = tid >> 5;      // wave id 0..7
  const int hi   = lane >> 4;     // lane half
  const int ln   = lane & 15;
  const int mrow0 = blockIdx.x * MTILE;
  const int c     = blockIdx.y;   // vocab split id

  // cooperative Q tile load (32 rows x 512 bf16 = 2048 uint4)
  {
    const uint4* src = (const uint4*)(Qbf + (size_t)mrow0 * D_TEXT);
    uint4* dst = (uint4*)qt;
#pragma unroll
    for (int l = 0; l < 8; ++l) dst[tid + l * 256] = src[tid + l * 256];
  }
  if (tid < MTILE) dsum[tid] = 0.f;
  __syncthreads();

  v8f Olo[4], Ohi[4];
#pragma unroll
  for (int i = 0; i < 4; ++i) { Olo[i] = V8F_ZERO; Ohi[i] = V8F_ZERO; }
  float dacc0[8] = {0.f,0.f,0.f,0.f,0.f,0.f,0.f,0.f};
  float dacc1[8] = {0.f,0.f,0.f,0.f,0.f,0.f,0.f,0.f};

  const unsigned short* qrow0 = qt + ln * D_TEXT + hi * 8;          // rows 0..15
  const unsigned short* qrow1 = qt + (16 + ln) * D_TEXT + hi * 8;   // rows 16..31

  for (int t = c; t < NTILES; t += NSPLIT) {
    const int vbase = t * 128;

    // ---- phase 1: S(32x16) = Q(32x512) . K_chunk^T -----------------------
    v8f s0 = V8F_ZERO, s1 = V8F_ZERO;
    const unsigned short* krow =
        Kbf + (size_t)(vbase + w * 16 + ln) * D_TEXT + hi * 8;
    if (t + NSPLIT < NTILES)
      __builtin_prefetch(Kbf + (size_t)(vbase + NSPLIT * 128 + w * 16 + ln) * D_TEXT, 0, 1);

    FragBF kb[2];
    kb[0].q[0] = *(const uint4*)(krow + 0);
    kb[0].q[1] = *(const uint4*)(krow + 16);
#pragma unroll
    for (int step = 0; step < 16; ++step) {
      const int d0  = step * 32;
      const int cur = step & 1, nxt = cur ^ 1;
      if (step < 15) {                       // double-buffer next K frag
        kb[nxt].q[0] = *(const uint4*)(krow + d0 + 32);
        kb[nxt].q[1] = *(const uint4*)(krow + d0 + 48);
      }
      FragBF a0, a1;
      a0.q[0] = *(const uint4*)(qrow0 + d0);
      a0.q[1] = *(const uint4*)(qrow0 + d0 + 16);
      a1.q[0] = *(const uint4*)(qrow1 + d0);
      a1.q[1] = *(const uint4*)(qrow1 + d0 + 16);
      s0 = __builtin_amdgcn_wmma_f32_16x16x32_bf16(false, a0.v, false, kb[cur].v,
                                                   (short)0, s0, false, false);
      s1 = __builtin_amdgcn_wmma_f32_16x16x32_bf16(false, a1.v, false, kb[cur].v,
                                                   (short)0, s1, false, false);
    }

    // ---- exp(score / T): cos in [-1,1] -> exp <= e^10, no max needed -----
#pragma unroll
    for (int r = 0; r < 8; ++r) {
      float p0 = exp2f(s0[r] * VQ_SCALE);
      float p1 = exp2f(s1[r] * VQ_SCALE);
      dacc0[r] += p0;
      dacc1[r] += p1;
      ptile[(r + 8 * hi) * 128 + w * 16 + ln]      = f2bf(p0);
      ptile[(16 + r + 8 * hi) * 128 + w * 16 + ln] = f2bf(p1);
    }
    __syncthreads();

    // ---- phase 2: O[:, w*64 .. w*64+64) += P(32x128) . V(128x64) ---------
    FragBF pl[4], ph[4];                     // hoisted P frags (both halves)
#pragma unroll
    for (int ks = 0; ks < 4; ++ks) {
      const int kb2 = ks * 32;
      pl[ks].q[0] = *(const uint4*)(ptile + ln * 128 + kb2 + hi * 8);
      pl[ks].q[1] = *(const uint4*)(ptile + ln * 128 + kb2 + 16 + hi * 8);
      ph[ks].q[0] = *(const uint4*)(ptile + (16 + ln) * 128 + kb2 + hi * 8);
      ph[ks].q[1] = *(const uint4*)(ptile + (16 + ln) * 128 + kb2 + 16 + hi * 8);
    }
    const int dbase = w * 64;
#pragma unroll
    for (int nb = 0; nb < 4; ++nb) {
      const unsigned short* vtrow =
          Vt + (size_t)(dbase + nb * 16 + ln) * VOCAB + vbase + hi * 8;
#pragma unroll
      for (int ks = 0; ks < 4; ++ks) {
        FragBF b;
        b.q[0] = *(const uint4*)(vtrow + ks * 32);
        b.q[1] = *(const uint4*)(vtrow + ks * 32 + 16);
        Olo[nb] = __builtin_amdgcn_wmma_f32_16x16x32_bf16(false, pl[ks].v, false, b.v,
                                                          (short)0, Olo[nb], false, false);
        Ohi[nb] = __builtin_amdgcn_wmma_f32_16x16x32_bf16(false, ph[ks].v, false, b.v,
                                                          (short)0, Ohi[nb], false, false);
      }
    }
    __syncthreads();
  }

  // ---- denominator: reduce across the 16 lanes of each half, then waves ---
#pragma unroll
  for (int off = 1; off < 16; off <<= 1)
#pragma unroll
    for (int r = 0; r < 8; ++r) {
      dacc0[r] += __shfl_xor(dacc0[r], off, 32);
      dacc1[r] += __shfl_xor(dacc1[r], off, 32);
    }
  if (ln == 0) {
#pragma unroll
    for (int r = 0; r < 8; ++r) {
      atomicAdd(&dsum[r + 8 * hi], dacc0[r]);
      atomicAdd(&dsum[16 + r + 8 * hi], dacc1[r]);
    }
  }

  // ---- numerator partials: C/D layout (vgpr r, lane) -> (m=r+8*hi, n=ln) --
  float* nbase = numer + ((size_t)c * NROWS + mrow0) * D_TEXT + w * 64;
#pragma unroll
  for (int nb = 0; nb < 4; ++nb)
#pragma unroll
    for (int r = 0; r < 8; ++r) {
      nbase[(size_t)(r + 8 * hi) * D_TEXT + nb * 16 + ln]      = Olo[nb][r];
      nbase[(size_t)(16 + r + 8 * hi) * D_TEXT + nb * 16 + ln] = Ohi[nb][r];
    }

  __syncthreads();
  if (tid < MTILE) denom[(size_t)c * NROWS + mrow0 + tid] = dsum[tid];
}

// ---------------- kernel 7: combine vocab-split partials -------------------
__global__ void combine_kernel(const float* __restrict__ numer,
                               const float* __restrict__ denom,
                               float* __restrict__ out) {
  const size_t i = (size_t)blockIdx.x * blockDim.x + threadIdx.x;
  if (i >= (size_t)NROWS * D_TEXT) return;
  const size_t m = i / D_TEXT;
  float n = 0.f, d = 0.f;
#pragma unroll
  for (int c = 0; c < NSPLIT; ++c) {
    n += numer[(size_t)c * NROWS * D_TEXT + i];
    d += denom[(size_t)c * NROWS + m];
  }
  out[i] = n / d;
}

// ---------------------------------------------------------------------------
extern "C" void kernel_launch(void* const* d_in, const int* in_sizes, int n_in,
                              void* d_out, int out_size, void* d_ws, size_t ws_size,
                              hipStream_t stream) {
  (void)in_sizes; (void)n_in; (void)out_size; (void)ws_size;
  const float* audio = (const float*)d_in[0];   // [2048, 768]
  const float* W     = (const float*)d_in[1];   // [768, 512]
  const float* bias  = (const float*)d_in[2];   // [512]
  const float* gamma = (const float*)d_in[3];   // [512]
  const float* beta  = (const float*)d_in[4];   // [512]
  const float* emb   = (const float*)d_in[5];   // [49408, 512]

  char* ws = (char*)d_ws;
  float*          proj   = (float*)(ws);                    //  4.00 MB
  float*          mean   = (float*)(ws + 4194304);          //  2 KB
  float*          invstd = (float*)(ws + 4196352);          //  2 KB
  unsigned short* Qbf    = (unsigned short*)(ws + 4198400); //  2.00 MB
  unsigned short* Kbf    = (unsigned short*)(ws + 6295552); // 50.59 MB
  unsigned short* Vt     = (unsigned short*)(ws + 56889344);// 50.59 MB
  float*          numer  = (float*)(ws + 107483136);        // 16.78 MB
  float*          denom  = (float*)(ws + 124260352);        // 32 KB
  float*          out    = (float*)d_out;                   // [2048, 512]

  proj_kernel   <<<NROWS, 256, 0, stream>>>(audio, W, bias, proj);
  bnstats_kernel<<<D_TEXT, 256, 0, stream>>>(proj, mean, invstd);
  qprep_kernel  <<<NROWS, 256, 0, stream>>>(proj, mean, invstd, gamma, beta, Qbf);
  kprep_kernel  <<<VOCAB, 256, 0, stream>>>(emb, Kbf);
  vt_kernel     <<<dim3(VOCAB / 64, D_TEXT / 64), 256, 0, stream>>>(emb, Vt);
  attn_kernel   <<<dim3(NROWS / MTILE, NSPLIT), 256, 0, stream>>>(Qbf, Kbf, Vt, numer, denom);
  combine_kernel<<<(NROWS * D_TEXT + 255) / 256, 256, 0, stream>>>(numer, denom, out);
}
